// Model_V10_Alpha_72138270703812
// MI455X (gfx1250) — matrix operand order repaired and checked by reading, not compiled, not run on hardware
//
#include <hip/hip_runtime.h>
#include <hip/hip_bf16.h>
#include <math.h>

// ---------------------------------------------------------------------------
// CDNA5 (gfx1250) transformer forward. All GEMMs use v_wmma_f32_16x16x32_f16
// (f16 A/B, f32 accumulate). fp32 tensors in HBM, converted to f16 in LDS.
// v2: 128x128 block tiles (8 WMMA / wave / K-chunk) + register-prefetch
//     software pipeline to hide global latency behind the matrix pipe.
// ---------------------------------------------------------------------------

typedef __attribute__((ext_vector_type(16))) _Float16 v16h;
typedef __attribute__((ext_vector_type(8)))  float    v8f;

#define BM 128
#define BK 32
#define LDSK 34   // padded row stride (halves) -> 68B rows, conflict-avoiding

// --- fragment loaders, layouts per ISA 7.12.2 (wave32) ----------------------
// A 16x32 f16: lanes 0-15 row M=lane, VGPR v<4: K=2v,2v+1 (+8 if lane>=16),
//              VGPR v>=4: K=16+2(v-4),.. (+8 if lane>=16)
__device__ __forceinline__ v16h frag_a(const _Float16* sA, int row0, int lane) {
  const int hl = lane >> 4, r = lane & 15;
  const _Float16* base = sA + (row0 + r) * LDSK;
  union { v16h h; unsigned int u[8]; } f;
#pragma unroll
  for (int vv = 0; vv < 8; ++vv) {
    const int k = (vv < 4) ? (2 * vv + 8 * hl) : (16 + 2 * (vv - 4) + 8 * hl);
    f.u[vv] = *(const unsigned int*)(base + k);
  }
  return f.h;
}

// B 32x16 f16 stored in LDS as [N][K]: lanes 0-15 col N=lane, same K order
// as frag_a (VGPR v<4: K=2v,2v+1 (+8 if lane>=16), v>=4: K=16+2(v-4),..).
__device__ __forceinline__ v16h frag_b(const _Float16* sB, int n0, int lane) {
  const int hl = lane >> 4, r = lane & 15;
  const _Float16* base = sB + (n0 + r) * LDSK;
  union { v16h h; unsigned int u[8]; } f;
#pragma unroll
  for (int vv = 0; vv < 8; ++vv) {
    const int k = (vv < 4) ? (2 * vv + 8 * hl) : (16 + 2 * (vv - 4) + 8 * hl);
    f.u[vv] = *(const unsigned int*)(base + k);
  }
  return f.h;
}

// --- WMMA GEMM: C = alpha * A(MxK) * B(KxN) [+bias][relu][+Res] -------------
// WNT: N-subtiles per wave; block = 8 waves as 4(M) x 2(N); BN = 32*WNT.
// TRANSB: B given row-major [N][K] (i.e. compute A * B^T of stored matrix).
// Batched over grid.z: z -> (zb = z/HDIV, zh = z%HDIV) pointer offsets.
// All M,N,K used here are exact multiples of the tile sizes (no guards).
template <int WNT, bool TRANSB, bool BIAS, bool RELU, bool RES>
__global__ __launch_bounds__(256) void gemm_wmma_k(
    const float* __restrict__ A, long long lda, long long sAb, long long sAh,
    const float* __restrict__ B, long long ldb, long long sBb, long long sBh,
    const float* __restrict__ bias,
    const float* __restrict__ Res, long long ldr,
    float* __restrict__ C, long long ldc, long long sCb, long long sCh,
    int K, int HDIV, float alpha) {
  constexpr int BN = 32 * WNT;     // 64 or 128
  constexpr int PB = BN / 32;      // B-tile float4 passes per thread
  __shared__ _Float16 sA[BM * LDSK];
  __shared__ _Float16 sB[BN * LDSK];

  const int z = blockIdx.z;
  const int zb = z / HDIV, zh = z % HDIV;
  A += (long long)zb * sAb + (long long)zh * sAh;
  B += (long long)zb * sBb + (long long)zh * sBh;
  C += (long long)zb * sCb + (long long)zh * sCh;

  const int tid  = threadIdx.x;
  const int lane = tid & 31;
  const int wave = tid >> 5;
  const int wm = wave & 3;   // wave tile row  (32 rows each)
  const int wn = wave >> 2;  // wave tile col  (16*WNT cols each)
  const long long m0 = (long long)blockIdx.y * BM;
  const long long n0 = (long long)blockIdx.x * BN;

  const int c0a = (tid & 7) << 2;             // k-offset for A / TRANSB-B loads
  const int nnb = (tid & (BN / 4 - 1)) << 2;  // n-offset for [K][N]-B loads

  float4 pa[4], pb[PB];

  // prefetch one K-chunk of A and B into registers
  auto ld_ab = [&](int k0) {
#pragma unroll
    for (int p = 0; p < 4; ++p) {
      const int r = (tid >> 3) + (p << 5);
      pa[p] = *(const float4*)(A + (m0 + r) * lda + (k0 + c0a));
    }
    if (TRANSB) {
#pragma unroll
      for (int p = 0; p < PB; ++p) {
        const int n = (tid >> 3) + (p << 5);
        pb[p] = *(const float4*)(B + (n0 + n) * ldb + (k0 + c0a));
      }
    } else {
#pragma unroll
      for (int p = 0; p < PB; ++p) {
        const int kk = (tid / (BN / 4)) + p * (1024 / BN);
        pb[p] = *(const float4*)(B + (long long)(k0 + kk) * ldb + (n0 + nnb));
      }
    }
  };

  // convert fp32 regs -> f16 LDS tiles (sB stored [N][K])
  auto st_ab = [&]() {
#pragma unroll
    for (int p = 0; p < 4; ++p) {
      const int r = (tid >> 3) + (p << 5);
      _Float16* dst = sA + r * LDSK + c0a;
      dst[0] = (_Float16)pa[p].x; dst[1] = (_Float16)pa[p].y;
      dst[2] = (_Float16)pa[p].z; dst[3] = (_Float16)pa[p].w;
    }
    if (TRANSB) {
#pragma unroll
      for (int p = 0; p < PB; ++p) {
        const int n = (tid >> 3) + (p << 5);
        _Float16* dst = sB + n * LDSK + c0a;
        dst[0] = (_Float16)pb[p].x; dst[1] = (_Float16)pb[p].y;
        dst[2] = (_Float16)pb[p].z; dst[3] = (_Float16)pb[p].w;
      }
    } else {
#pragma unroll
      for (int p = 0; p < PB; ++p) {
        const int kk = (tid / (BN / 4)) + p * (1024 / BN);
        sB[(nnb + 0) * LDSK + kk] = (_Float16)pb[p].x;
        sB[(nnb + 1) * LDSK + kk] = (_Float16)pb[p].y;
        sB[(nnb + 2) * LDSK + kk] = (_Float16)pb[p].z;
        sB[(nnb + 3) * LDSK + kk] = (_Float16)pb[p].w;
      }
    }
  };

  v8f acc[2][WNT] = {};

  ld_ab(0);
  for (int k0 = 0; k0 < K; k0 += BK) {
    st_ab();
    __syncthreads();
    if (k0 + BK < K) ld_ab(k0 + BK);  // global loads in flight during WMMAs

    const v16h af0 = frag_a(sA, wm * 32,      lane);
    const v16h af1 = frag_a(sA, wm * 32 + 16, lane);
    v16h bf[WNT];
#pragma unroll
    for (int j = 0; j < WNT; ++j)
      bf[j] = frag_b(sB, wn * (16 * WNT) + j * 16, lane);
#pragma unroll
    for (int j = 0; j < WNT; ++j) {
      acc[0][j] = __builtin_amdgcn_wmma_f32_16x16x32_f16(
          false, af0, false, bf[j], (short)0, acc[0][j], false, false);
      acc[1][j] = __builtin_amdgcn_wmma_f32_16x16x32_f16(
          false, af1, false, bf[j], (short)0, acc[1][j], false, false);
    }
    __syncthreads();
  }

  // epilogue: C element of VGPR v at (row = v + 8*hl, col = lane&15)
  const int hl = lane >> 4, r = lane & 15;
#pragma unroll
  for (int i = 0; i < 2; ++i) {
#pragma unroll
    for (int j = 0; j < WNT; ++j) {
      const long long gn = n0 + wn * (16 * WNT) + j * 16 + r;
#pragma unroll
      for (int vv = 0; vv < 8; ++vv) {
        const long long gm = m0 + wm * 32 + i * 16 + vv + 8 * hl;
        float val = alpha * acc[i][j][vv];
        if (BIAS) val += bias[gn];
        if (RELU) val = fmaxf(val, 0.0f);
        if (RES)  val += Res[gm * ldr + gn];
        C[gm * ldc + gn] = val;
      }
    }
  }
}

// --- embedding + positional encoding ---------------------------------------
__global__ __launch_bounds__(256) void embed_k(
    const int* __restrict__ tok, const float* __restrict__ emb,
    const float* __restrict__ pos, float* __restrict__ out,
    int S, int d, float scale) {
  const long long idx = (long long)blockIdx.x * 256 + threadIdx.x;
  const long long t = idx / d;
  const int i = (int)(idx % d);
  const int id = tok[t];
  out[idx] = emb[(long long)id * d + i] * scale + pos[(t % S) * (long long)d + i];
}

// --- layernorm over d=512, gamma=p[0..511], beta=p[512..1023] ---------------
__global__ __launch_bounds__(256) void layernorm_k(
    const float* __restrict__ x, const float* __restrict__ p,
    float* __restrict__ out) {
  const int d = 512;
  const long long row = blockIdx.x;
  const float* xr = x + row * (long long)d;
  const int t = threadIdx.x;
  const float a = xr[t], b = xr[t + 256];
  __shared__ float red[256];
  red[t] = a + b;
  __syncthreads();
  for (int s = 128; s > 0; s >>= 1) { if (t < s) red[t] += red[t + s]; __syncthreads(); }
  const float mu = red[0] * (1.0f / d);
  __syncthreads();
  const float da = a - mu, db = b - mu;
  red[t] = da * da + db * db;
  __syncthreads();
  for (int s = 128; s > 0; s >>= 1) { if (t < s) red[t] += red[t + s]; __syncthreads(); }
  const float rs = rsqrtf(red[0] * (1.0f / d) + 1e-5f);
  float* o = out + row * (long long)d;
  o[t]       = p[t]       * da * rs + p[512 + t];
  o[t + 256] = p[t + 256] * db * rs + p[512 + t + 256];
}

// --- masked softmax over attention scores (in place), S=512, H=8 ------------
__global__ __launch_bounds__(256) void attn_softmax_k(
    float* __restrict__ scores, const int* __restrict__ qtok,
    const int* __restrict__ ktok, int causal) {
  const int S = 512;
  const int qi = blockIdx.x;
  const int z = blockIdx.y;
  const int b = z >> 3;  // H = 8
  float* row = scores + ((long long)z * S + qi) * (long long)S;
  const bool qv = (qtok[b * S + qi] != 1);  // PAD == 1
  const int t = threadIdx.x;
  float v[2];
#pragma unroll
  for (int j = 0; j < 2; ++j) {
    const int c = t + j * 256;
    const bool ok = qv && (ktok[b * S + c] != 1) && (!causal || c <= qi);
    v[j] = ok ? row[c] : -1e9f;
  }
  __shared__ float red[256];
  red[t] = fmaxf(v[0], v[1]);
  __syncthreads();
  for (int s = 128; s > 0; s >>= 1) { if (t < s) red[t] = fmaxf(red[t], red[t + s]); __syncthreads(); }
  const float m = red[0];
  __syncthreads();
  const float e0 = __expf(v[0] - m), e1 = __expf(v[1] - m);
  red[t] = e0 + e1;
  __syncthreads();
  for (int s = 128; s > 0; s >>= 1) { if (t < s) red[t] += red[t + s]; __syncthreads(); }
  const float inv = 1.0f / red[0];
  row[t] = e0 * inv;
  row[t + 256] = e1 * inv;
}

// --- row log-softmax over V=32000 (in place) --------------------------------
__global__ __launch_bounds__(256) void logsoftmax_k(float* __restrict__ logits) {
  const int V = 32000;
  float* r = logits + (long long)blockIdx.x * V;
  const int t = threadIdx.x;
  float m = -1e30f;
  for (int c = t; c < V; c += 256) m = fmaxf(m, r[c]);
  __shared__ float red[256];
  red[t] = m;
  __syncthreads();
  for (int s = 128; s > 0; s >>= 1) { if (t < s) red[t] = fmaxf(red[t], red[t + s]); __syncthreads(); }
  m = red[0];
  __syncthreads();
  float sum = 0.0f;
  for (int c = t; c < V; c += 256) sum += __expf(r[c] - m);
  red[t] = sum;
  __syncthreads();
  for (int s = 128; s > 0; s >>= 1) { if (t < s) red[t] += red[t + s]; __syncthreads(); }
  const float lse = m + __logf(red[0]);
  for (int c = t; c < V; c += 256) r[c] = r[c] - lse;
}

// ---------------------------------------------------------------------------
extern "C" void kernel_launch(void* const* d_in, const int* in_sizes, int n_in,
                              void* d_out, int out_size, void* d_ws, size_t ws_size,
                              hipStream_t stream) {
  (void)in_sizes; (void)n_in; (void)out_size; (void)ws_size;
  const int Bz = 4, S = 512, D = 512, DFF = 2048, V = 32000, L = 6, H = 8;
  const long long T = (long long)Bz * S;  // 2048 rows

  const int*   src         = (const int*)d_in[0];
  const int*   tgt         = (const int*)d_in[1];
  const float* src_emb     = (const float*)d_in[2];
  const float* tgt_emb     = (const float*)d_in[3];
  const float* pos_enc     = (const float*)d_in[4];
  const float* enc_attn_w  = (const float*)d_in[5];
  const float* enc_attn_b  = (const float*)d_in[6];
  const float* enc_ln      = (const float*)d_in[7];
  const float* enc_ff1_w   = (const float*)d_in[8];
  const float* enc_ff1_b   = (const float*)d_in[9];
  const float* enc_ff2_w   = (const float*)d_in[10];
  const float* enc_ff2_b   = (const float*)d_in[11];
  const float* enc_finalln = (const float*)d_in[12];
  const float* dec_sattn_w = (const float*)d_in[13];
  const float* dec_sattn_b = (const float*)d_in[14];
  const float* dec_cattn_w = (const float*)d_in[15];
  const float* dec_cattn_b = (const float*)d_in[16];
  const float* dec_ln      = (const float*)d_in[17];
  const float* dec_ff1_w   = (const float*)d_in[18];
  const float* dec_ff1_b   = (const float*)d_in[19];
  const float* dec_ff2_w   = (const float*)d_in[20];
  const float* dec_ff2_b   = (const float*)d_in[21];
  const float* dec_finalln = (const float*)d_in[22];
  const float* gen_w       = (const float*)d_in[23];
  const float* gen_b       = (const float*)d_in[24];

  // workspace layout (fp32): ~80 MB
  float* x      = (float*)d_ws;
  float* xn     = x      + T * D;
  float* q      = xn     + T * D;
  float* kbuf   = q      + T * D;
  float* vbuf   = kbuf   + T * D;
  float* attno  = vbuf   + T * D;
  float* encout = attno  + T * D;
  float* ffh    = encout + T * D;
  float* scores = ffh    + T * DFF;  // [B,H,S,S] = 32*512*512 floats

  float* logits = (float*)d_out;
  float* decout = logits + T * (long long)V;

  const float emb_scale = 22.62741699796952f;  // sqrt(512)
  const float inv_sqrt_dk = 0.125f;            // 1/sqrt(64)
  const long long SD = (long long)S * D;       // batch stride of activations
  const long long SS = (long long)S * S;       // per-(b,h) score stride

  // ---- shared attention block: LN'd input -> q,k,v -> softmax(pv) -> proj+res
  auto attention = [&](const float* qin, const float* kvin,
                       const float* W, const float* Bv,
                       const int* qtk, const int* ktk, int causal,
                       float* resid_io) {
    const long long dd = (long long)D * D;
    // q / k / v projections  (M=2048, N=512, K=512)
    gemm_wmma_k<4, false, true, false, false><<<dim3(4, 16, 1), 256, 0, stream>>>(
        qin, D, 0, 0, W + 0 * dd, D, 0, 0, Bv + 0 * D, nullptr, 0,
        q, D, 0, 0, 512, 1, 1.0f);
    gemm_wmma_k<4, false, true, false, false><<<dim3(4, 16, 1), 256, 0, stream>>>(
        kvin, D, 0, 0, W + 1 * dd, D, 0, 0, Bv + 1 * D, nullptr, 0,
        kbuf, D, 0, 0, 512, 1, 1.0f);
    gemm_wmma_k<4, false, true, false, false><<<dim3(4, 16, 1), 256, 0, stream>>>(
        kvin, D, 0, 0, W + 2 * dd, D, 0, 0, Bv + 2 * D, nullptr, 0,
        vbuf, D, 0, 0, 512, 1, 1.0f);
    // scores = q . k^T / sqrt(dk)   (per (b,h): M=512, N=512, K=64; TRANSB)
    gemm_wmma_k<4, true, false, false, false><<<dim3(4, 4, 32), 256, 0, stream>>>(
        q, D, SD, 64, kbuf, D, SD, 64, nullptr, nullptr, 0,
        scores, S, (long long)H * SS, SS, 64, H, inv_sqrt_dk);
    // mask + softmax in place
    attn_softmax_k<<<dim3(S, Bz * H), 256, 0, stream>>>(scores, qtk, ktk, causal);
    // out = probs . v   (per (b,h): M=512, N=64, K=512) -> BN=64 variant
    gemm_wmma_k<2, false, false, false, false><<<dim3(1, 4, 32), 256, 0, stream>>>(
        scores, S, (long long)H * SS, SS, vbuf, D, SD, 64, nullptr, nullptr, 0,
        attno, D, SD, 64, 512, H, 1.0f);
    // out projection + residual   (M=2048, N=512, K=512)
    gemm_wmma_k<4, false, true, false, true><<<dim3(4, 16, 1), 256, 0, stream>>>(
        attno, D, 0, 0, W + 3 * dd, D, 0, 0, Bv + 3 * D, resid_io, D,
        resid_io, D, 0, 0, 512, 1, 1.0f);
  };

  auto ffn = [&](const float* in, const float* w1, const float* b1,
                 const float* w2, const float* b2, float* resid_io) {
    gemm_wmma_k<4, false, true, true, false><<<dim3(16, 16, 1), 256, 0, stream>>>(
        in, D, 0, 0, w1, DFF, 0, 0, b1, nullptr, 0, ffh, DFF, 0, 0, 512, 1, 1.0f);
    gemm_wmma_k<4, false, true, false, true><<<dim3(4, 16, 1), 256, 0, stream>>>(
        ffh, DFF, 0, 0, w2, D, 0, 0, b2, resid_io, D, resid_io, D, 0, 0, 2048, 1, 1.0f);
  };

  // =================== encoder ===================
  embed_k<<<(unsigned)(T * D / 256), 256, 0, stream>>>(src, src_emb, pos_enc, x, S, D, emb_scale);
  for (int i = 0; i < L; ++i) {
    const float* W  = enc_attn_w + (long long)i * 4 * D * D;
    const float* Bv = enc_attn_b + (long long)i * 4 * D;
    layernorm_k<<<(unsigned)T, 256, 0, stream>>>(x, enc_ln + (long long)(i * 2 + 0) * 2 * D, xn);
    attention(xn, xn, W, Bv, src, src, 0, x);
    layernorm_k<<<(unsigned)T, 256, 0, stream>>>(x, enc_ln + (long long)(i * 2 + 1) * 2 * D, xn);
    ffn(xn, enc_ff1_w + (long long)i * D * DFF, enc_ff1_b + (long long)i * DFF,
        enc_ff2_w + (long long)i * DFF * D, enc_ff2_b + (long long)i * D, x);
  }
  layernorm_k<<<(unsigned)T, 256, 0, stream>>>(x, enc_finalln, encout);

  // =================== decoder ===================
  embed_k<<<(unsigned)(T * D / 256), 256, 0, stream>>>(tgt, tgt_emb, pos_enc, x, S, D, emb_scale);
  for (int i = 0; i < L; ++i) {
    const float* Ws  = dec_sattn_w + (long long)i * 4 * D * D;
    const float* Bs  = dec_sattn_b + (long long)i * 4 * D;
    const float* Wc  = dec_cattn_w + (long long)i * 4 * D * D;
    const float* Bc  = dec_cattn_b + (long long)i * 4 * D;
    // masked self-attention (causal + pad)
    layernorm_k<<<(unsigned)T, 256, 0, stream>>>(x, dec_ln + (long long)(i * 3 + 0) * 2 * D, xn);
    attention(xn, xn, Ws, Bs, tgt, tgt, 1, x);
    // cross attention: q from decoder stream, k/v from encoder output
    layernorm_k<<<(unsigned)T, 256, 0, stream>>>(x, dec_ln + (long long)(i * 3 + 1) * 2 * D, xn);
    attention(xn, encout, Wc, Bc, tgt, src, 0, x);
    // ffn
    layernorm_k<<<(unsigned)T, 256, 0, stream>>>(x, dec_ln + (long long)(i * 3 + 2) * 2 * D, xn);
    ffn(xn, dec_ff1_w + (long long)i * D * DFF, dec_ff1_b + (long long)i * DFF,
        dec_ff2_w + (long long)i * DFF * D, dec_ff2_b + (long long)i * D, x);
  }
  layernorm_k<<<(unsigned)T, 256, 0, stream>>>(x, dec_finalln, decout);

  // =================== generator ===================
  // logits = dec_out @ gen_w + gen_b   (M=2048, N=32000, K=512)
  gemm_wmma_k<4, false, true, false, false><<<dim3(250, 16, 1), 256, 0, stream>>>(
      decout, D, 0, 0, gen_w, V, 0, 0, gen_b, nullptr, 0,
      logits, V, 0, 0, 512, 1, 1.0f);
  logsoftmax_k<<<(unsigned)T, 256, 0, stream>>>(logits);
}
